// AngularProtoLoss_15006615733567
// MI455X (gfx1250) — matrix operand order repaired and checked by reading, not compiled, not run on hardware
//
#include <hip/hip_runtime.h>

typedef __attribute__((ext_vector_type(16))) _Float16 v16h;
typedef __attribute__((ext_vector_type(8)))  _Float16 v8h;
typedef __attribute__((ext_vector_type(8)))  float    v8f;

#define NROWS 8192
#define NSPK  8192
#define DIM   256
#define MARGIN 0.2f
#define L2EPS  1e-12f

// ---------------------------------------------------------------------------
// Kernel 1: L2-normalize each embedding row; emit fp32 (for proto sums) and
// fp16 (for the WMMA GEMM A-matrix) copies.
// ---------------------------------------------------------------------------
__global__ __launch_bounds__(256) void normalize_rows(
    const float* __restrict__ x, float* __restrict__ e32,
    _Float16* __restrict__ e16) {
  __shared__ float red[8];
  __shared__ float scale_s;
  const int row = blockIdx.x;
  const int d   = threadIdx.x;
  float v  = x[(size_t)row * DIM + d];
  float ss = v * v;
  #pragma unroll
  for (int m = 16; m >= 1; m >>= 1) ss += __shfl_xor(ss, m, 32);
  if ((threadIdx.x & 31) == 0) red[threadIdx.x >> 5] = ss;
  __syncthreads();
  if (threadIdx.x == 0) {
    float t = 0.f;
    #pragma unroll
    for (int i = 0; i < 8; ++i) t += red[i];
    scale_s = 1.0f / fmaxf(sqrtf(t), L2EPS);
  }
  __syncthreads();
  float e = v * scale_s;
  e32[(size_t)row * DIM + d] = e;
  e16[(size_t)row * DIM + d] = (_Float16)e;
}

// ---------------------------------------------------------------------------
// Kernel 2: build prototypes. One block per speaker; labels cached in LDS;
// fixed-order scan (deterministic, no float atomics). Then mean + L2-normalize
// the prototype row and emit fp16 (WMMA B-matrix).
// ---------------------------------------------------------------------------
__global__ __launch_bounds__(256) void build_protos(
    const float* __restrict__ e32, const int* __restrict__ labels,
    _Float16* __restrict__ p16) {
  __shared__ int   lab_s[NROWS];   // 32 KB
  __shared__ float red[8];
  __shared__ float scale_s;
  const int s = blockIdx.x;
  const int d = threadIdx.x;
  for (int t = threadIdx.x; t < NROWS; t += 256) lab_s[t] = labels[t];
  __syncthreads();

  float acc = 0.f;
  int   cnt = 0;
  for (int t = 0; t < NROWS; ++t) {
    if (lab_s[t] == s) {               // wave-uniform branch, rarely taken
      acc += e32[(size_t)t * DIM + d];
      ++cnt;
    }
  }
  float p = acc / fmaxf((float)cnt, 1.0f);  // zero rows stay zero

  float ss = p * p;
  #pragma unroll
  for (int m = 16; m >= 1; m >>= 1) ss += __shfl_xor(ss, m, 32);
  if ((threadIdx.x & 31) == 0) red[threadIdx.x >> 5] = ss;
  __syncthreads();
  if (threadIdx.x == 0) {
    float t = 0.f;
    #pragma unroll
    for (int i = 0; i < 8; ++i) t += red[i];
    scale_s = 1.0f / fmaxf(sqrtf(t), L2EPS);
  }
  __syncthreads();
  p16[(size_t)s * DIM + d] = (_Float16)(p * scale_s);
}

// ---------------------------------------------------------------------------
// Kernel 3: fused GEMM + row-max + positive-gather.
// Block = 4 waves; all waves share the 16-row A strip (in registers) and own
// disjoint quarters of the 8192 columns. The 2x-unrolled ping-pong loop gives
// the compiler two independent v_wmma_f32_16x16x32_f16 accumulation chains to
// interleave, and global_prefetch (2 compute phases ahead) pulls the next B
// tiles from L2 into WGP$ so the demand b128 loads are near-L0 latency.
// ---------------------------------------------------------------------------
__global__ __launch_bounds__(128) void gemm_fused(
    const _Float16* __restrict__ e16, const _Float16* __restrict__ p16,
    const int* __restrict__ labels, float* __restrict__ loss) {
  const int lane = threadIdx.x & 31;
  const int wave = threadIdx.x >> 5;   // 0..3
  const int hl   = lane >> 4;          // half-wave select
  const int sub  = lane & 15;
  const int row0 = blockIdx.x * 16;

  // ---- A strip: rows row0..row0+15, K=0..255. 16-bit A 16x32 layout
  // (ISA 7.12.2): lane group hl, halves 0..7 -> K = kk + 8*hl + t,
  // halves 8..15 -> K = kk + 16 + 8*hl + t. Both contiguous b128 loads.
  v16h A[8];
  {
    const _Float16* arow = e16 + (size_t)(row0 + sub) * DIM;
    #pragma unroll
    for (int c = 0; c < 8; ++c) {
      const int kk = c * 32;
      v8h lo = *(const v8h*)(arow + kk + hl * 8);
      v8h hi = *(const v8h*)(arow + kk + 16 + hl * 8);
      v16h a;
      #pragma unroll
      for (int t = 0; t < 8; ++t) { a[t] = lo[t]; a[t + 8] = hi[t]; }
      A[c] = a;
    }
  }

  // C/D layout: lane holds col N = sub; VGPR v holds row M = v + 8*hl.
  int labv[8];
  #pragma unroll
  for (int v = 0; v < 8; ++v) labv[v] = labels[row0 + v + 8 * hl];

  float rowmax[8], pos[8];
  #pragma unroll
  for (int v = 0; v < 8; ++v) { rowmax[v] = -1e30f; pos[v] = -1e30f; }

  // B tile 32x16 f16 (ISA 7.12.2): lane = col N = sub, halves 0..7 ->
  // K = kk + 8*hl + t, halves 8..15 -> K = kk + 16 + 8*hl + t (matches A).
  auto loadB = [&](int j0, v16h* B) {
    const _Float16* brow = p16 + (size_t)(j0 + sub) * DIM + hl * 8;
    #pragma unroll
    for (int c = 0; c < 8; ++c) {
      v8h lo = *(const v8h*)(brow + c * 32);
      v8h hi = *(const v8h*)(brow + c * 32 + 16);
      v16h b;
      #pragma unroll
      for (int t = 0; t < 8; ++t) { b[t] = lo[t]; b[t + 8] = hi[t]; }
      B[c] = b;
    }
  };

  // Prefetch one whole 16x256 B tile into WGP$: per row (lane sub), the two
  // 128B lines at byte offsets hl*256 and hl*256+128 -> all 512 B covered
  // across the 32 lanes with 2 global_prefetch instructions.
  auto prefetchB = [&](int j0) {
    const char* base = (const char*)(p16 + (size_t)(j0 + sub) * DIM) + hl * 256;
    __builtin_prefetch(base, 0, 3);
    __builtin_prefetch(base + 128, 0, 3);
  };

  auto compute = [&](int j0, const v16h* B) {
    v8f acc = {};
    #pragma unroll
    for (int c = 0; c < 8; ++c) {
      acc = __builtin_amdgcn_wmma_f32_16x16x32_f16(
          /*neg_a=*/false, A[c], /*neg_b=*/false, B[c],
          /*c_mod=*/(short)0, acc, /*reuse_a=*/false, /*reuse_b=*/false);
    }
    const int col = j0 + sub;
    #pragma unroll
    for (int v = 0; v < 8; ++v) {
      const int row = row0 + v + 8 * hl;
      const float val = acc[v];
      const float vm  = (col == row) ? (val - MARGIN) : val;  // diagonal margin
      rowmax[v] = fmaxf(rowmax[v], vm);
      if (col == labv[v]) pos[v] = fmaxf(pos[v], val);        // raw positive sim
    }
  };

  // Per-wave column tiles: j = wave*16 + k*64, k = 0..127. Ping-pong B
  // buffers, 2x unrolled; prefetch runs two compute phases ahead. Tail
  // prefetch/load indices are clamped to a valid tile and never consumed.
  const int jbase = wave * 16;
  v16h B0[8], B1[8];
  prefetchB(jbase);
  prefetchB(jbase + 64);
  loadB(jbase, B0);
  loadB(jbase + 64, B1);
  for (int k = 0; k < 128; k += 2) {
    const int ja = jbase + k * 64;
    const int jc = (k + 2 < 128) ? ja + 128 : jbase;
    const int jd = (k + 3 < 128) ? ja + 192 : jbase;
    const int je = (k + 4 < 128) ? ja + 256 : jbase;
    const int jf = (k + 5 < 128) ? ja + 320 : jbase;
    prefetchB(je);           // warm WGP$ two phases ahead
    prefetchB(jf);
    compute(ja, B0);
    loadB(jc, B0);           // in flight during compute of B1's tile
    compute(ja + 64, B1);
    loadB(jd, B1);           // in flight during next compute of B0's tile
  }

  // Reduce across the 16 lanes sharing each row set.
  #pragma unroll
  for (int m = 8; m >= 1; m >>= 1) {
    #pragma unroll
    for (int v = 0; v < 8; ++v) {
      rowmax[v] = fmaxf(rowmax[v], __shfl_xor(rowmax[v], m, 16));
      pos[v]    = fmaxf(pos[v],    __shfl_xor(pos[v],    m, 16));
    }
  }

  // Combine the 4 column-partition waves through LDS.
  __shared__ float smax[4][16];
  __shared__ float spos[4][16];
  if (sub == 0) {
    #pragma unroll
    for (int v = 0; v < 8; ++v) {
      smax[wave][v + 8 * hl] = rowmax[v];
      spos[wave][v + 8 * hl] = pos[v];
    }
  }
  __syncthreads();
  if (threadIdx.x < 16) {
    float m = -1e30f, p = -1e30f;
    #pragma unroll
    for (int w = 0; w < 4; ++w) {
      m = fmaxf(m, smax[w][threadIdx.x]);
      p = fmaxf(p, spos[w][threadIdx.x]);
    }
    const float t = m - p + MARGIN;
    loss[row0 + threadIdx.x] = t > 0.f ? t : 0.f;
  }
}

// ---------------------------------------------------------------------------
// Kernel 4: deterministic fixed-order mean of the 8192 per-row losses.
// ---------------------------------------------------------------------------
__global__ __launch_bounds__(256) void reduce_loss(
    const float* __restrict__ loss, float* __restrict__ out) {
  __shared__ float red[256];
  float s = 0.f;
  for (int i = threadIdx.x; i < NROWS; i += 256) s += loss[i];
  red[threadIdx.x] = s;
  __syncthreads();
  for (int stride = 128; stride >= 1; stride >>= 1) {
    if (threadIdx.x < stride) red[threadIdx.x] += red[threadIdx.x + stride];
    __syncthreads();
  }
  if (threadIdx.x == 0) out[0] = red[0] / (float)NROWS;
}

// ---------------------------------------------------------------------------
extern "C" void kernel_launch(void* const* d_in, const int* in_sizes, int n_in,
                              void* d_out, int out_size, void* d_ws, size_t ws_size,
                              hipStream_t stream) {
  const float* embeddings = (const float*)d_in[0];   // [8192, 256] f32
  const int*   labels     = (const int*)d_in[1];     // [8192] i32
  float*       out        = (float*)d_out;           // scalar f32

  char* ws = (char*)d_ws;
  float*    e32  = (float*)ws;                                   // 8 MB
  _Float16* e16  = (_Float16*)(ws + (size_t)NROWS * DIM * 4);    // 4 MB
  _Float16* p16  = (_Float16*)(ws + (size_t)NROWS * DIM * 6);    // 4 MB
  float*    loss = (float*)(ws + (size_t)NROWS * DIM * 8);       // 32 KB

  normalize_rows<<<NROWS, 256, 0, stream>>>(embeddings, e32, e16);
  build_protos<<<NSPK, 256, 0, stream>>>(e32, labels, p16);
  gemm_fused<<<NROWS / 16, 128, 0, stream>>>(e16, p16, labels, loss);
  reduce_loss<<<1, 256, 0, stream>>>(loss, out);
}
